// ModularTreeMLPPredictor_42649025249413
// MI455X (gfx1250) — matrix operand order repaired and checked by reading, not compiled, not run on hardware
//
#include <hip/hip_runtime.h>

// Tree-of-MLPs, level-by-level (11 sequential launches on `stream`).
// f16 WMMA (v_wmma_f32_16x16x32_f16) w/ f32 accumulate. Per wave: 2 x 16-row
// M-tiles sharing B-tiles in registers. Transposed f16 weights + W3 staged in
// LDS once per block. Bank-conflict-free LDS strides (72 halves = 36 dwords).

#define TD 11
#define TF 32
#define TH 64
#define TB 512
#define TN 2047
#define XS 72      // LDS row stride in halves
#define WAVES 4    // 128-thread blocks, 4 waves, 32 rows per wave

typedef _Float16 half_t;
typedef _Float16 v16h __attribute__((ext_vector_type(16)));
typedef _Float16 v8h  __attribute__((ext_vector_type(8)));
typedef _Float16 v2h  __attribute__((ext_vector_type(2)));
typedef float    v8f  __attribute__((ext_vector_type(8)));

__device__ __forceinline__ v16h cat16(v8h lo, v8h hi) {
  return __builtin_shufflevector(lo, hi, 0,1,2,3,4,5,6,7,8,9,10,11,12,13,14,15);
}

__global__ void __launch_bounds__(128)
tree_mlp_level(const float* __restrict__ features,
               const float* __restrict__ treatment,
               const float* __restrict__ W1, const float* __restrict__ b1,
               const float* __restrict__ W2, const float* __restrict__ b2,
               const float* __restrict__ W3, const float* __restrict__ b3,
               float* __restrict__ out, int l, int n, int leaf)
{
  // LDS: w1t [64][72]h (9216B) | w2t (9216B) | w3s 64f (256B) | 4 waves x
  //      (xh [32][72]h 4608B + hh [32][72]h 4608B)  => 55552B total
  __shared__ __align__(16) unsigned char smem[18688 + WAVES * 9216];
  half_t* w1t = (half_t*)smem;
  half_t* w2t = (half_t*)(smem + 9216);
  float*  w3s = (float*)(smem + 18432);
  const int tid  = threadIdx.x;
  const int lane = tid & 31;
  const int wave = tid >> 5;
  half_t* xh = (half_t*)(smem + 18688 + wave * 9216);  // x, then reused for h2
  half_t* hh = xh + 32 * XS;                           // h1

  // ---- stage transposed f16 weights + W3 (once per block, coalesced)
  for (int idx = tid; idx < TH * TH; idx += 128) {
    int k = idx >> 6, col = idx & 63;
    w1t[col * XS + k] = (half_t)((k < TF + 3) ? W1[k * TH + col] : 0.0f);
    w2t[col * XS + k] = (half_t)W2[k * TH + col];
  }
  if (tid < TH) w3s[tid] = W3[tid];

  const int rows       = TB << l;
  const int start      = n - 1;
  const int childStart = 2 * n - 1;
  const int tileBase   = (blockIdx.x * WAVES + wave) * 32;

  // ---- stage features: 2 rows/iter, float2 loads -> packed f16 -> b32 store
  for (int i = 0; i < 16; ++i) {
    int rr = i * 2 + (lane >> 4);
    int r = tileBase + rr; if (r >= rows) r = rows - 1;
    int b = r >> l, node = r & (n - 1);
    const float2 f2 = *(const float2*)(features +
        ((size_t)b * TN + start + node) * TF + (lane & 15) * 2);
    v2h hv = { (half_t)f2.x, (half_t)f2.y };
    *(v2h*)(xh + rr * XS + (lane & 15) * 2) = hv;
  }
  // ---- extras: one row per lane (left, right, treatment, zero-pad K->64)
  {
    int r = tileBase + lane; if (r >= rows) r = rows - 1;
    int b = r >> l, node = r & (n - 1);
    float lv = 0.0f, rv = 0.0f;
    if (!leaf) {
      size_t cb = (size_t)b * TN + childStart + 2 * node;
      lv = out[cb]; rv = out[cb + 1];
    }
    half_t* xr = xh + lane * XS;
    xr[32] = (half_t)lv;
    xr[33] = (half_t)rv;
    xr[34] = (half_t)treatment[b];
    xr[35] = (half_t)0.0f;
    v2h z2 = {}; *(v2h*)(xr + 36) = z2; *(v2h*)(xr + 38) = z2;
    v8h z8 = {}; *(v8h*)(xr + 40) = z8; *(v8h*)(xr + 48) = z8; *(v8h*)(xr + 56) = z8;
  }
  __syncthreads();

  const int m   = lane & 15;          // row within M-tile (A) / col (B,C)
  const int hiA = (lane >> 4) << 3;   // A K offset (0/8); also C/D M offset
  const int hiB = (lane >> 4) << 3;   // B K offset (0/8)

  v8f c0[4], c1[4];

  // ================= layer 1 =================
  #pragma unroll
  for (int nn = 0; nn < 4; ++nn) {
    float bv = b1[nn * 16 + m];
    #pragma unroll
    for (int i = 0; i < 8; ++i) { c0[nn][i] = bv; c1[nn][i] = bv; }
  }
  #pragma unroll
  for (int s = 0; s < 2; ++s) {
    const half_t* pa0 = xh + m        * XS + s * 32 + hiA;
    const half_t* pa1 = xh + (16 + m) * XS + s * 32 + hiA;
    v16h a0 = cat16(*(const v8h*)pa0, *(const v8h*)(pa0 + 16));
    v16h a1 = cat16(*(const v8h*)pa1, *(const v8h*)(pa1 + 16));
    #pragma unroll
    for (int nn = 0; nn < 4; ++nn) {
      const half_t* pb = w1t + (nn * 16 + m) * XS + s * 32 + hiB;
      v16h bm = cat16(*(const v8h*)pb, *(const v8h*)(pb + 16));
      c0[nn] = __builtin_amdgcn_wmma_f32_16x16x32_f16(false, a0, false, bm,
                                                      (short)0, c0[nn], false, false);
      c1[nn] = __builtin_amdgcn_wmma_f32_16x16x32_f16(false, a1, false, bm,
                                                      (short)0, c1[nn], false, false);
    }
  }
  #pragma unroll
  for (int nn = 0; nn < 4; ++nn)
    #pragma unroll
    for (int i = 0; i < 8; ++i) {
      hh[(i + hiA) * XS + nn * 16 + m]      = (half_t)fmaxf(c0[nn][i], 0.0f);
      hh[(16 + i + hiA) * XS + nn * 16 + m] = (half_t)fmaxf(c1[nn][i], 0.0f);
    }
  __syncthreads();

  // ================= layer 2 =================
  #pragma unroll
  for (int nn = 0; nn < 4; ++nn) {
    float bv = b2[nn * 16 + m];
    #pragma unroll
    for (int i = 0; i < 8; ++i) { c0[nn][i] = bv; c1[nn][i] = bv; }
  }
  #pragma unroll
  for (int s = 0; s < 2; ++s) {
    const half_t* pa0 = hh + m        * XS + s * 32 + hiA;
    const half_t* pa1 = hh + (16 + m) * XS + s * 32 + hiA;
    v16h a0 = cat16(*(const v8h*)pa0, *(const v8h*)(pa0 + 16));
    v16h a1 = cat16(*(const v8h*)pa1, *(const v8h*)(pa1 + 16));
    #pragma unroll
    for (int nn = 0; nn < 4; ++nn) {
      const half_t* pb = w2t + (nn * 16 + m) * XS + s * 32 + hiB;
      v16h bm = cat16(*(const v8h*)pb, *(const v8h*)(pb + 16));
      c0[nn] = __builtin_amdgcn_wmma_f32_16x16x32_f16(false, a0, false, bm,
                                                      (short)0, c0[nn], false, false);
      c1[nn] = __builtin_amdgcn_wmma_f32_16x16x32_f16(false, a1, false, bm,
                                                      (short)0, c1[nn], false, false);
    }
  }
  // ReLU -> reuse xh as h2
  #pragma unroll
  for (int nn = 0; nn < 4; ++nn)
    #pragma unroll
    for (int i = 0; i < 8; ++i) {
      xh[(i + hiA) * XS + nn * 16 + m]      = (half_t)fmaxf(c0[nn][i], 0.0f);
      xh[(16 + i + hiA) * XS + nn * 16 + m] = (half_t)fmaxf(c1[nn][i], 0.0f);
    }
  __syncthreads();

  // ================= layer 3: one row per lane =================
  {
    int r = tileBase + lane;
    if (r < rows) {
      float acc = b3[0];
      const half_t* hr = xh + lane * XS;
      #pragma unroll
      for (int j = 0; j < 8; ++j) {
        v8h hv = *(const v8h*)(hr + j * 8);
        #pragma unroll
        for (int e = 0; e < 8; ++e)
          acc += (float)hv[e] * w3s[j * 8 + e];
      }
      int b = r >> l, node = r & (n - 1);
      out[(size_t)b * TN + start + node] = acc;
    }
  }
}

extern "C" void kernel_launch(void* const* d_in, const int* in_sizes, int n_in,
                              void* d_out, int out_size, void* d_ws, size_t ws_size,
                              hipStream_t stream) {
  (void)in_sizes; (void)n_in; (void)out_size; (void)d_ws; (void)ws_size;
  const float* features  = (const float*)d_in[0];
  const float* treatment = (const float*)d_in[1];
  const float* W1 = (const float*)d_in[2];
  const float* b1 = (const float*)d_in[3];
  const float* W2 = (const float*)d_in[4];
  const float* b2 = (const float*)d_in[5];
  const float* W3 = (const float*)d_in[6];
  const float* b3 = (const float*)d_in[7];
  float* out = (float*)d_out;

  for (int l = TD - 1; l >= 0; --l) {
    int n    = 1 << l;
    int rows = TB * n;
    int blocks = (rows + 32 * WAVES - 1) / (32 * WAVES);
    tree_mlp_level<<<blocks, 32 * WAVES, 0, stream>>>(
        features, treatment,
        W1 + (size_t)l * (TF + 3) * TH, b1 + (size_t)l * TH,
        W2 + (size_t)l * TH * TH,       b2 + (size_t)l * TH,
        W3 + (size_t)l * TH,            b3 + l,
        out, l, n, (l == TD - 1) ? 1 : 0);
  }
}
